// Codebook_33689723469824
// MI455X (gfx1250) — matrix operand order repaired and checked by reading, not compiled, not run on hardware
//
#include <hip/hip_runtime.h>
#include <hip/hip_bf16.h>
#include <math.h>

typedef __attribute__((ext_vector_type(16))) __bf16 v16bf;
typedef __attribute__((ext_vector_type(8)))  __bf16 v8bf;
typedef __attribute__((ext_vector_type(4)))  __bf16 v4bf;
typedef __attribute__((ext_vector_type(8)))  float  v8f;
typedef __attribute__((ext_vector_type(4)))  float  f4;

#define N_CODES 1024
#define DIM     256
#define NVEC    65536          // 4*16*32*32 flat vectors
#define SPAT    16384          // 16*32*32 spatial per batch
#define ROWS    64             // z rows staged per block
#define THREADS 128            // 4 waves of 32

// ws layout: [0..1023] esq f32 | [1024..2047] counts u32 | [2048] loss f32
//            | byte offset 16384: Ebf (1024*256 bf16 = 524288 B)
#define WS_EBF_OFF   16384
#define WS_BYTES_BF  (WS_EBF_OFF + N_CODES * DIM * 2)

// ---------------------------------------------------------------------------
// Kernel 0a: per-code squared norms, zero counters/loss (ws poisoned 0xAA)
// ---------------------------------------------------------------------------
__global__ void vq_prep(const float* __restrict__ E,
                        float* __restrict__ esq,
                        unsigned* __restrict__ counts,
                        float* __restrict__ loss) {
    int c = blockIdx.x * blockDim.x + threadIdx.x;
    if (c < N_CODES) {
        const float* row = E + (size_t)c * DIM;
        float s = 0.f;
        for (int k = 0; k < DIM; k += 4) {
            f4 v = *(const f4*)(row + k);
            s += v.x * v.x + v.y * v.y + v.z * v.z + v.w * v.w;
        }
        esq[c]    = s;
        counts[c] = 0u;
    }
    if (c == 0) *loss = 0.f;
}

// ---------------------------------------------------------------------------
// Kernel 0b: one-shot fp32 -> bf16 conversion of the embedding table
// ---------------------------------------------------------------------------
__global__ void vq_cvt(const float* __restrict__ E, __bf16* __restrict__ Ebf) {
    int i = blockIdx.x * blockDim.x + threadIdx.x;   // one f4 / 4 outputs each
    f4 v = ((const f4*)E)[i];
    v4bf o;
    o[0] = (__bf16)v.x; o[1] = (__bf16)v.y;
    o[2] = (__bf16)v.z; o[3] = (__bf16)v.w;
    ((v4bf*)Ebf)[i] = o;
}

// ---------------------------------------------------------------------------
// Kernel 1: WMMA score GEMM + argmin + bincount + gather/scatter + loss part
// ---------------------------------------------------------------------------
template <bool BF>
__global__ void __launch_bounds__(THREADS)
vq_main(const float* __restrict__ z,
        const float* __restrict__ E,
        const __bf16* __restrict__ Ebf,
        const float* __restrict__ esq,
        unsigned* __restrict__ counts,
        float* __restrict__ loss,
        float* __restrict__ out_emb,   // [B,C,T,H,W] quantized (== embeddings_st)
        float* __restrict__ out_enc)   // [B,T,H,W] indices as float
{
    __shared__ __attribute__((aligned(16))) __bf16 zt[ROWS][DIM + 8];
    __shared__ int   sidx[ROWS];
    __shared__ float sred[THREADS / 32];

    const int tid  = threadIdx.x;
    const int base = blockIdx.x * ROWS;    // first flat vector of this tile

    // ---- stage 64 rows x 256 channels of z into LDS as bf16 ----
    // thread handles (row, channel-pair): global loads coalesced over rows,
    // LDS stores are packed b32.
    for (int i = 0; i < (ROWS * DIM / 2) / THREADS; ++i) {
        int linear = i * THREADS + tid;
        int r  = linear & (ROWS - 1);      // row in tile -> consecutive w
        int cp = linear >> 6;              // channel pair
        int n = base + r;
        int b = n >> 14;                   // /SPAT
        int s = n & (SPAT - 1);
        size_t za = (size_t)b * ((size_t)DIM * SPAT) + (size_t)(2 * cp) * SPAT + s;
        float v0 = z[za];
        float v1 = z[za + SPAT];
        v4bf* dst = (v4bf*)nullptr;        // silence unused in some paths
        (void)dst;
        __bf16 b0 = (__bf16)v0, b1 = (__bf16)v1;
        // one packed 32-bit LDS store
        *(__attribute__((ext_vector_type(2))) __bf16*)&zt[r][2 * cp] =
            (__attribute__((ext_vector_type(2))) __bf16){b0, b1};
    }
    __syncthreads();

    const int wave    = tid >> 5;
    const int lane    = tid & 31;
    const int half    = lane >> 4;         // 0: lanes 0-15, 1: lanes 16-31
    const int col     = lane & 15;
    const int rowbase = wave * 16;         // this wave's 16 rows within tile

    // ---- build the 8 A fragments: per fragment two aligned b128 LDS loads ---
    // A layout: lane holds row M = lane%16; element e -> K = 16*half + e
    v16bf afr[8];
    {
        const __bf16* zr = &zt[rowbase + col][0];
        #pragma unroll
        for (int kk = 0; kk < 8; ++kk) {
            v8bf lo = *(const v8bf*)(zr + kk * 32 + 16 * half);       // K: +0..7
            v8bf hi = *(const v8bf*)(zr + kk * 32 + 16 * half + 8);   // K: +8..15
            afr[kk] = __builtin_shufflevector(lo, hi,
                        0, 1, 2, 3, 4, 5, 6, 7, 8, 9, 10, 11, 12, 13, 14, 15);
        }
    }

    // ---- sweep 64 code tiles; track per-lane argmin of (-2*score + |e|^2) ---
    float bestv[8];
    int   besti[8];
    #pragma unroll
    for (int r = 0; r < 8; ++r) { bestv[r] = 3.4e38f; besti[r] = 0; }

    #pragma unroll 2
    for (int ct = 0; ct < 64; ++ct) {
        const int code = ct * 16 + col;                    // this lane's column
        v8f acc = {};
        #pragma unroll
        for (int kk = 0; kk < 8; ++kk) {
            // B layout: lane holds col N = lane%16; K = kk*32 + 16*half + e
            v16bf bf;
            if constexpr (BF) {
                const __bf16* p = Ebf + (size_t)code * DIM + kk * 32 + 16 * half;
                bf = *(const v16bf*)p;                     // 32 B -> 2x b128
            } else {
                const float* p = E + (size_t)code * DIM + kk * 32 + 16 * half;
                f4 v0 = *(const f4*)(p + 0);
                f4 v1 = *(const f4*)(p + 4);
                f4 v2 = *(const f4*)(p + 8);
                f4 v3 = *(const f4*)(p + 12);
                bf[0]  = (__bf16)v0.x; bf[1]  = (__bf16)v0.y;
                bf[2]  = (__bf16)v0.z; bf[3]  = (__bf16)v0.w;
                bf[4]  = (__bf16)v1.x; bf[5]  = (__bf16)v1.y;
                bf[6]  = (__bf16)v1.z; bf[7]  = (__bf16)v1.w;
                bf[8]  = (__bf16)v2.x; bf[9]  = (__bf16)v2.y;
                bf[10] = (__bf16)v2.z; bf[11] = (__bf16)v2.w;
                bf[12] = (__bf16)v3.x; bf[13] = (__bf16)v3.y;
                bf[14] = (__bf16)v3.z; bf[15] = (__bf16)v3.w;
            }
            acc = __builtin_amdgcn_wmma_f32_16x16x32_bf16(
                    false, afr[kk], false, bf, (short)0, acc, false, false);
        }
        const float eq = esq[code];
        #pragma unroll
        for (int r = 0; r < 8; ++r) {
            float d = eq - 2.0f * acc[r];   // ||z||^2 constant per row: omitted
            if (d < bestv[r]) { bestv[r] = d; besti[r] = code; }
        }
    }

    // ---- cross-lane argmin within each 16-lane half (XOR shuffles) ----
    #pragma unroll
    for (int r = 0; r < 8; ++r) {
        float bv = bestv[r];
        int   bi = besti[r];
        #pragma unroll
        for (int m = 1; m < 16; m <<= 1) {
            float ov = __shfl_xor(bv, m, 32);
            int   oi = __shfl_xor(bi, m, 32);
            if (ov < bv || (ov == bv && oi < bi)) { bv = ov; bi = oi; }
        }
        bestv[r] = bv; besti[r] = bi;
    }

    // element r of C/D holds row M = r + 8*half; lane col==0 publishes results
    if (col == 0) {
        #pragma unroll
        for (int r = 0; r < 8; ++r) {
            int row  = rowbase + 8 * half + r;
            int code = besti[r];
            sidx[row] = code;
            out_enc[base + row] = (float)code;
            atomicAdd(&counts[code], 1u);
        }
    }
    __syncthreads();

    // ---- gather E[idx] -> emb output (coalesced over w), loss partial ----
    float lsum = 0.f;
    for (int i = 0; i < (ROWS * DIM) / THREADS; ++i) {
        int linear = i * THREADS + tid;
        int r = linear & (ROWS - 1);
        int c = linear >> 6;
        int n = base + r;
        int b = n >> 14;
        int s = n & (SPAT - 1);
        size_t za = (size_t)b * ((size_t)DIM * SPAT) + (size_t)c * SPAT + s;
        float ev = E[(size_t)sidx[r] * DIM + c];
        out_emb[za] = ev;                      // embeddings_st == emb numerically
        float d = z[za] - ev;
        lsum += d * d;
    }
    // wave shuffle reduce, then tiny LDS combine
    #pragma unroll
    for (int m = 16; m > 0; m >>= 1) lsum += __shfl_xor(lsum, m, 32);
    if (lane == 0) sred[wave] = lsum;
    __syncthreads();
    if (tid == 0) {
        float t = 0.f;
        #pragma unroll
        for (int i = 0; i < THREADS / 32; ++i) t += sred[i];
        // mean over 4*256*16384 = 16777216 elements, times 0.25
        atomicAdd(loss, t * (0.25f / 16777216.0f));
    }
}

// ---------------------------------------------------------------------------
// Kernel 2: perplexity from counts + scalar writes
// ---------------------------------------------------------------------------
__global__ void vq_finalize(const unsigned* __restrict__ counts,
                            const float* __restrict__ loss,
                            float* __restrict__ out_scal) {
    __shared__ float sred[N_CODES];
    int c = threadIdx.x;
    float p = (float)counts[c] * (1.0f / (float)NVEC);
    sred[c] = -p * __logf(p + 1e-10f);
    __syncthreads();
    for (int off = N_CODES / 2; off > 0; off >>= 1) {
        if (c < off) sred[c] += sred[c + off];
        __syncthreads();
    }
    if (c == 0) {
        out_scal[0] = *loss;            // commitment_loss
        out_scal[1] = __expf(sred[0]);  // perplexity
    }
}

// ---------------------------------------------------------------------------
extern "C" void kernel_launch(void* const* d_in, const int* in_sizes, int n_in,
                              void* d_out, int out_size, void* d_ws, size_t ws_size,
                              hipStream_t stream) {
    const float* z = (const float*)d_in[0];   // [4,256,16,32,32]
    const float* E = (const float*)d_in[1];   // [1024,256]

    float* out      = (float*)d_out;
    float* out_emb  = out;                               // 4194304 floats
    float* out_enc  = out + 4194304;                     // 65536 floats
    float* out_scal = out + 4194304 + 65536;             // loss, perplexity

    float*    esq    = (float*)d_ws;                     // [1024]
    unsigned* counts = (unsigned*)d_ws + 1024;           // [1024]
    float*    loss   = (float*)d_ws + 2048;              // [1]
    __bf16*   Ebf    = (__bf16*)((char*)d_ws + WS_EBF_OFF);

    vq_prep<<<4, 256, 0, stream>>>(E, esq, counts, loss);

    if (ws_size >= (size_t)WS_BYTES_BF) {
        // pre-converted bf16 table path (L2-resident, no hot-loop cvts)
        vq_cvt<<<(N_CODES * DIM / 4) / 256, 256, 0, stream>>>(E, Ebf);
        vq_main<true><<<NVEC / ROWS, THREADS, 0, stream>>>(
            z, E, Ebf, esq, counts, loss, out_emb, out_enc);
    } else {
        vq_main<false><<<NVEC / ROWS, THREADS, 0, stream>>>(
            z, E, Ebf, esq, counts, loss, out_emb, out_enc);
    }

    vq_finalize<<<1, N_CODES, 0, stream>>>(counts, loss, out_scal);
}
